// Bases_7078106104328
// MI455X (gfx1250) — compile-verified, matrix operand order checked by reading, not run on hardware
//
#include <hip/hip_runtime.h>
#include <hip/hip_bf16.h>

// ---------------- problem constants ----------------
#define N_ATOMS   8000
#define N_EDGES   200000
#define N_TRIP    1000000
#define EMB_ATOM  256
#define EMB_EDGE  512
#define NUM_RADIAL 128
#define NUM_SPH   7
#define MTILES    (N_EDGES / 16)      // 12500
#define KT_EDGE   20                  // 640 / 32 k-tiles for edge GEMM
#define NT_EDGE   32                  // 512 / 16 n-tiles
#define KT_SMALL  4                   // 128 / 32
#define NT_SMALL  10                  // 160 / 16  (112 cbf + 16 + 16 + 16)

typedef __attribute__((ext_vector_type(16))) __bf16 bf16x16;
typedef __attribute__((ext_vector_type(8)))  float  f32x8;
typedef __attribute__((ext_vector_type(4)))  unsigned int u32x4;
typedef __attribute__((ext_vector_type(8)))  int  i32x8;
typedef __attribute__((ext_vector_type(4)))  int  i32x4;

#if defined(__has_builtin)
#  if __has_builtin(__builtin_amdgcn_tensor_load_to_lds)
#    define USE_TDM 1
#  endif
#endif
#ifndef USE_TDM
#  define USE_TDM 0
#endif

// pack two floats into one dword of bf16 pair (lo = a, hi = b)
__device__ __forceinline__ unsigned pack2bf(float a, float b) {
    union { __bf16 h; unsigned short u; } x, y;
    x.h = (__bf16)a; y.h = (__bf16)b;
    return ((unsigned)y.u << 16) | (unsigned)x.u;
}

// K offset within a 32-wide k-tile for lane `lane`, dword j (WMMA bf16 A/B layout):
// lanes 0-15 own K {0..7, 16..23}; lanes 16-31 own K {8..15, 24..31}; dword j holds K=kk, kk+1
__device__ __forceinline__ int kk_of(int lane, int j) {
    return ((lane >> 4) << 3) + 2 * j + ((j >= 4) ? 8 : 0);
}

#if USE_TDM
// Tensor Data Mover: DMA one contiguous 32 KB slice (4096 x 8B elements) from
// global memory into LDS.  D# per CDNA5 ISA 8.3/8.4: group0 = {count=1, lds_addr,
// global_addr, type=2}; group1 = 1-D tile: data_size=8B, tensor_dim0=tile_dim0=4096.
// This toolchain exposes the 6-arg builtin:
//   (u32x4 g0, i32x8 g1, i32x4 g2, i32x4 g3, i32x8 extra, i32 cpol)
__device__ __forceinline__ void tdm_copy_32k(const unsigned* __restrict__ src,
                                             unsigned lds_byte_off) {
    unsigned long long ga = (unsigned long long)(size_t)src;
    u32x4 g0;
    g0.x = 1u;                                            // count=1, user mode
    g0.y = lds_byte_off;                                  // D#.lds_addr
    g0.z = (unsigned)ga;                                  // global_addr[31:0]
    g0.w = (unsigned)((ga >> 32) & 0x01FFFFFFu)           // global_addr[56:32]
         | 0x80000000u;                                   // type=2 ("image")
    const unsigned NEL = 4096;                            // 8-byte elements
    i32x8 g1;
    g1.s0 = (int)(3u << 16);                              // data_size = 3 (8 bytes)
    g1.s1 = (int)((NEL & 0xFFFFu) << 16);                 // tensor_dim0 lo16
    g1.s2 = (int)((NEL >> 16) | (1u << 16));              // tensor_dim0 hi | tensor_dim1=1
    g1.s3 = (int)(NEL << 16);                             // tile_dim0 = 4096
    g1.s4 = 1;                                            // tile_dim1 = 1
    g1.s5 = (int)NEL;                                     // tensor_dim0_stride
    g1.s6 = 0; g1.s7 = 0;
    i32x4 z4 = {0, 0, 0, 0};
    i32x8 z8 = {0, 0, 0, 0, 0, 0, 0, 0};
    __builtin_amdgcn_tensor_load_to_lds(g0, g1, z4, z4, z8, 0);
}
#endif

// ---------------------------------------------------------------------------
// Kernel 1: radial basis, written directly in WMMA-A-fragment-packed bf16.
// radA[mtile][kt(4)][lane(32)][j(8)]  (dwords)
// ---------------------------------------------------------------------------
__global__ __launch_bounds__(256) void rad_pack_kernel(const float* __restrict__ distance,
                                                       unsigned* __restrict__ radA) {
    int g = blockIdx.x * 256 + threadIdx.x;      // (mtile, kt, lane)
    if (g >= MTILES * KT_SMALL * 32) return;
    int lane  = g & 31;
    int kt    = (g >> 5) & 3;
    int mtile = g >> 7;
    int row   = lane & 15;

    float d  = distance[mtile * 16 + row];
    float ds = d * (1.0f / 12.0f);
    float ds2 = ds * ds, ds4 = ds2 * ds2, ds5 = ds4 * ds;
    float env = (ds < 1.0f) ? (1.0f - 21.0f * ds5 + 35.0f * ds5 * ds - 15.0f * ds5 * ds2) : 0.0f;
    const float coeff = -0.5f * 127.0f * 127.0f;   // -0.5 / (1/127)^2

    unsigned w[8];
#pragma unroll
    for (int j = 0; j < 8; ++j) {
        int k0 = kt * 32 + kk_of(lane, j);
        float t0 = ds - (float)k0       * (1.0f / 127.0f);
        float t1 = ds - (float)(k0 + 1) * (1.0f / 127.0f);
        float v0 = env * __expf(coeff * t0 * t0);
        float v1 = env * __expf(coeff * t1 * t1);
        w[j] = pack2bf(v0, v1);
    }
    uint4* p = (uint4*)&radA[(size_t)g * 8];
    p[0] = make_uint4(w[0], w[1], w[2], w[3]);
    p[1] = make_uint4(w[4], w[5], w[6], w[7]);
}

// ---------------------------------------------------------------------------
// Kernel 2: pack W_edge (640x512 f32 row-major) into B-fragment bf16.
// BpkE[kt(20)][nt(32)][lane(32)][j(8)]
// ---------------------------------------------------------------------------
__global__ __launch_bounds__(256) void pack_wedge_kernel(const float* __restrict__ W,
                                                         unsigned* __restrict__ B) {
    int g = blockIdx.x * 256 + threadIdx.x;
    if (g >= KT_EDGE * NT_EDGE * 32) return;
    int lane = g & 31;
    int nt   = (g >> 5) & 31;
    int kt   = g >> 10;
    int col  = nt * 16 + (lane & 15);

    unsigned w[8];
#pragma unroll
    for (int j = 0; j < 8; ++j) {
        int k0 = kt * 32 + kk_of(lane, j);
        w[j] = pack2bf(W[(size_t)k0 * EMB_EDGE + col], W[(size_t)(k0 + 1) * EMB_EDGE + col]);
    }
    uint4* p = (uint4*)&B[(size_t)g * 8];
    p[0] = make_uint4(w[0], w[1], w[2], w[3]);
    p[1] = make_uint4(w[4], w[5], w[6], w[7]);
}

// ---------------------------------------------------------------------------
// Kernel 3: pack the 4 small weight matrices (K=128) into one N=160 B buffer.
// cols 0..111 = W_cbf_tint, 112..127 = W_rbf_tint, 128..143 = W_rbf_h, 144..159 = W_rbf_out
// ---------------------------------------------------------------------------
__global__ __launch_bounds__(256) void pack_wsmall_kernel(const float* __restrict__ Wcbf,
                                                          const float* __restrict__ Wtint,
                                                          const float* __restrict__ Wh,
                                                          const float* __restrict__ Wout,
                                                          unsigned* __restrict__ B) {
    int g = blockIdx.x * 256 + threadIdx.x;
    if (g >= KT_SMALL * NT_SMALL * 32) return;
    int lane = g & 31;
    int nt   = (g >> 5) % NT_SMALL;
    int kt   = (g >> 5) / NT_SMALL;
    int col  = nt * 16 + (lane & 15);

    unsigned w[8];
#pragma unroll
    for (int j = 0; j < 8; ++j) {
        int k0 = kt * 32 + kk_of(lane, j);
        float a, b;
        if (col < 112)      { a = Wcbf [(size_t)k0 * 112 + col];        b = Wcbf [(size_t)(k0 + 1) * 112 + col]; }
        else if (col < 128) { a = Wtint[(size_t)k0 * 16 + (col - 112)]; b = Wtint[(size_t)(k0 + 1) * 16 + (col - 112)]; }
        else if (col < 144) { a = Wh   [(size_t)k0 * 16 + (col - 128)]; b = Wh   [(size_t)(k0 + 1) * 16 + (col - 128)]; }
        else                { a = Wout [(size_t)k0 * 16 + (col - 144)]; b = Wout [(size_t)(k0 + 1) * 16 + (col - 144)]; }
        w[j] = pack2bf(a, b);
    }
    uint4* p = (uint4*)&B[(size_t)g * 8];
    p[0] = make_uint4(w[0], w[1], w[2], w[3]);
    p[1] = make_uint4(w[4], w[5], w[6], w[7]);
}

// ---------------------------------------------------------------------------
// Kernel 4: rad @ [Wcbf | Wtint | Wh | Wout]  via WMMA bf16. One M-tile per wave.
// ---------------------------------------------------------------------------
__global__ __launch_bounds__(128) void small_gemm_kernel(const unsigned* __restrict__ radA,
                                                         const unsigned* __restrict__ Bpk,
                                                         float* __restrict__ rW1,
                                                         float* __restrict__ out_atom,
                                                         float* __restrict__ out_output,
                                                         float* __restrict__ out_re) {
    int wave = threadIdx.x >> 5, lane = threadIdx.x & 31;
    int mtile = blockIdx.x * 4 + wave;
    if (mtile >= MTILES) return;

    f32x8 c[NT_SMALL] = {};
#pragma unroll
    for (int t = 0; t < KT_SMALL; ++t) {
        bf16x16 a = *(const bf16x16*)&radA[(size_t)(((mtile * 4 + t) * 32) + lane) * 8];
#pragma unroll
        for (int n = 0; n < NT_SMALL; ++n) {
            bf16x16 b = *(const bf16x16*)&Bpk[(size_t)(((t * NT_SMALL + n) * 32) + lane) * 8];
            c[n] = __builtin_amdgcn_wmma_f32_16x16x32_bf16(false, a, false, b,
                                                           (short)0, c[n], false, false);
        }
    }
    int rbase = mtile * 16 + ((lane >= 16) ? 8 : 0);
    int nlo   = lane & 15;
#pragma unroll
    for (int n = 0; n < 7; ++n)
#pragma unroll
        for (int v = 0; v < 8; ++v)
            rW1[(size_t)(rbase + v) * 112 + n * 16 + nlo] = c[n][v];
#pragma unroll
    for (int v = 0; v < 8; ++v) {
        out_re    [(size_t)(rbase + v) * 16 + nlo] = c[7][v];
        out_atom  [(size_t)(rbase + v) * 16 + nlo] = c[8][v];
        out_output[(size_t)(rbase + v) * 16 + nlo] = c[9][v];
    }
}

// ---------------------------------------------------------------------------
// Kernel 5: EdgeEmbedding  m = ScaledSiLU([h_s || h_t || rad] @ W_edge)
// 256 threads (8 waves), 32 rows per block.
//  - gathered h columns staged into LDS as packed bf16 A fragments
//  - W_edge k-slices (32 KB) DMA'd into a double-buffered LDS region by the
//    Tensor Data Mover (TENSORcnt), consumed by all 8 waves via ds_load_b128
//  - rad k-tiles come straight from the packed radA buffer
// ---------------------------------------------------------------------------
__global__ __launch_bounds__(256) void edge_embed_kernel(const float* __restrict__ h,
                                                         const int* __restrict__ idx_s,
                                                         const int* __restrict__ idx_t,
                                                         const unsigned* __restrict__ radA,
                                                         const unsigned* __restrict__ Bpk,
                                                         float* __restrict__ out_m) {
    __shared__ unsigned Alds[2 * 16 * 32 * 8];      // 32 KB: [msub][kt(16)][lane][j]
    __shared__ unsigned Blds[2][NT_EDGE * 32 * 8];  // 2 x 32 KB double buffer
    int blk = blockIdx.x;
    int edge_base = blk * 32;
    int tid = threadIdx.x;
    int wave = tid >> 5, lane = tid & 31;

#if USE_TDM
    // kick off DMA of k-tile 0 while we stage A
    if (wave == 0)
        tdm_copy_32k(Bpk, (unsigned)(size_t)&Blds[0][0]);
#endif

    // stage gathered h rows as bf16 A-fragments (8192 dwords, 32 per thread)
#pragma unroll
    for (int it = 0; it < 32; ++it) {
        int e    = tid + it * 256;
        int j    = e & 7;
        int la   = (e >> 3) & 31;
        int kt   = (e >> 8) & 15;
        int msub = e >> 12;
        int row  = la & 15;
        int edge = edge_base + msub * 16 + row;
        int idx  = (kt < 8) ? idx_s[edge] : idx_t[edge];
        int col  = ((kt & 7) << 5) + kk_of(la, j);
        const float* hp = h + (size_t)idx * EMB_ATOM + col;
        Alds[e] = pack2bf(hp[0], hp[1]);
    }
    __syncthreads();

    int msub = wave >> 2, ngrp = wave & 3;
    int mtile = blk * 2 + msub;

    f32x8 c[8] = {};
    for (int t = 0; t < KT_EDGE; ++t) {
#if USE_TDM
        if (wave == 0) {
            if (t + 1 < KT_EDGE) {
                tdm_copy_32k(Bpk + (size_t)(t + 1) * (NT_EDGE * 32 * 8),
                             (unsigned)(size_t)&Blds[(t + 1) & 1][0]);
                __builtin_amdgcn_s_wait_tensorcnt(1);   // slice t has landed
            } else {
                __builtin_amdgcn_s_wait_tensorcnt(0);
            }
        }
        __syncthreads();                                // slice t visible to all waves
        const unsigned* Bb = &Blds[t & 1][0];
#else
        __syncthreads();
        for (int it = tid; it < NT_EDGE * 32 * 8; it += 256)
            Blds[0][it] = Bpk[(size_t)t * (NT_EDGE * 32 * 8) + it];
        __syncthreads();
        const unsigned* Bb = &Blds[0][0];
#endif
        bf16x16 a;
        if (t < 16)
            a = *(const bf16x16*)&Alds[(size_t)(((msub * 16 + t) * 32) + lane) * 8];
        else
            a = *(const bf16x16*)&radA[(size_t)(((mtile * 4 + (t - 16)) * 32) + lane) * 8];
#pragma unroll
        for (int n = 0; n < 8; ++n) {
            bf16x16 b = *(const bf16x16*)&Bb[(size_t)(((ngrp * 8 + n) * 32) + lane) * 8];
            c[n] = __builtin_amdgcn_wmma_f32_16x16x32_bf16(false, a, false, b,
                                                           (short)0, c[n], false, false);
        }
        __syncthreads();   // all waves done with this buffer before it is re-filled
    }

    // fused ScaledSiLU epilogue
    int col0  = ngrp * 128;
    int rbase = edge_base + msub * 16 + ((lane >= 16) ? 8 : 0);
    int nlo   = lane & 15;
#pragma unroll
    for (int n = 0; n < 8; ++n)
#pragma unroll
        for (int v = 0; v < 8; ++v) {
            float x = c[n][v];
            float s = (x / (1.0f + __expf(-x))) * (1.0f / 0.6f);
            out_m[(size_t)(rbase + v) * EMB_EDGE + col0 + n * 16 + nlo] = s;
        }
}

// ---------------------------------------------------------------------------
// Kernel 6: triplet contraction  cir_e2e[t] = sum_l pref_l * P_l(cos) * rW1[to[t]][l*16+c]
// ---------------------------------------------------------------------------
__global__ __launch_bounds__(256) void triplet_kernel(const float* __restrict__ vec,
                                                      const int* __restrict__ trip_in,
                                                      const int* __restrict__ trip_out,
                                                      const float* __restrict__ rW1,
                                                      float* __restrict__ out_cir) {
    int t = blockIdx.x * 256 + threadIdx.x;
    if (t >= N_TRIP) return;
    int i = trip_in[t], o = trip_out[t];
    const float* vo = vec + (size_t)o * 3;
    const float* vi = vec + (size_t)i * 3;
    float x = vo[0] * vi[0] + vo[1] * vi[1] + vo[2] * vi[2];
    x = fminf(1.0f, fmaxf(-1.0f, x));

    float P[NUM_SPH];
    P[0] = 1.0f; P[1] = x;
#pragma unroll
    for (int l = 2; l < NUM_SPH; ++l)
        P[l] = ((float)(2 * l - 1) * x * P[l - 1] - (float)(l - 1) * P[l - 2]) * (1.0f / (float)l);

    const float pref[NUM_SPH] = {0.28209479f, 0.48860252f, 0.63078314f, 0.74635267f,
                                 0.84628440f, 0.93560258f, 1.01710724f};

    float acc[16];
#pragma unroll
    for (int c = 0; c < 16; ++c) acc[c] = 0.0f;

    const float* rw = rW1 + (size_t)o * 112;
#pragma unroll
    for (int l = 0; l < NUM_SPH; ++l) {
        float cl = pref[l] * P[l];
        const float* rr = rw + l * 16;
#pragma unroll
        for (int c = 0; c < 16; ++c) acc[c] += cl * rr[c];
    }
    float4* op = (float4*)(out_cir + (size_t)t * 16);
#pragma unroll
    for (int q = 0; q < 4; ++q)
        op[q] = make_float4(acc[4 * q], acc[4 * q + 1], acc[4 * q + 2], acc[4 * q + 3]);
}

// ---------------------------------------------------------------------------
extern "C" void kernel_launch(void* const* d_in, const int* in_sizes, int n_in,
                              void* d_out, int out_size, void* d_ws, size_t ws_size,
                              hipStream_t stream) {
    const float* h          = (const float*)d_in[0];
    const float* distance   = (const float*)d_in[1];
    const float* vector     = (const float*)d_in[2];
    const int*   edge_idx_s = (const int*)d_in[3];
    const int*   edge_idx_t = (const int*)d_in[4];
    const int*   trip_in    = (const int*)d_in[5];
    const int*   trip_out   = (const int*)d_in[6];
    const float* W_rbf_tint = (const float*)d_in[7];
    const float* W_cbf_tint = (const float*)d_in[8];
    const float* W_rbf_h    = (const float*)d_in[9];
    const float* W_rbf_out  = (const float*)d_in[10];
    const float* W_edge     = (const float*)d_in[11];

    float* out        = (float*)d_out;
    float* out_m      = out;                                        // (E, 512)
    float* out_atom   = out_m    + (size_t)N_EDGES * EMB_EDGE;      // (E, 16)
    float* out_output = out_atom + (size_t)N_EDGES * 16;            // (E, 16)
    float* out_re     = out_output + (size_t)N_EDGES * 16;          // (E, 16)
    float* out_cir    = out_re   + (size_t)N_EDGES * 16;            // (T, 16)

    // workspace layout
    char* ws = (char*)d_ws;
    unsigned* radA = (unsigned*)ws;                                 // 51,200,000 B
    unsigned* BpkE = (unsigned*)(ws + 51200000);                    //    655,360 B
    unsigned* BpkS = (unsigned*)(ws + 51200000 + 655360);           //     40,960 B
    float*    rW1  = (float*)   (ws + 51200000 + 655360 + 40960);   // 89,600,000 B

    rad_pack_kernel<<<(MTILES * KT_SMALL * 32 + 255) / 256, 256, 0, stream>>>(distance, radA);
    pack_wedge_kernel<<<(KT_EDGE * NT_EDGE * 32 + 255) / 256, 256, 0, stream>>>(W_edge, BpkE);
    pack_wsmall_kernel<<<(KT_SMALL * NT_SMALL * 32 + 255) / 256, 256, 0, stream>>>(
        W_cbf_tint, W_rbf_tint, W_rbf_h, W_rbf_out, BpkS);
    small_gemm_kernel<<<MTILES / 4, 128, 0, stream>>>(radA, BpkS, rW1,
                                                      out_atom, out_output, out_re);
    edge_embed_kernel<<<MTILES / 2, 256, 0, stream>>>(h, edge_idx_s, edge_idx_t,
                                                      radA, BpkE, out_m);
    triplet_kernel<<<(N_TRIP + 255) / 256, 256, 0, stream>>>(vector, trip_in, trip_out,
                                                             rW1, out_cir);
}
